// FineGrainedStudent_80685255622832
// MI455X (gfx1250) — matrix operand order repaired and checked by reading, not compiled, not run on hardware
//
#include <hip/hip_runtime.h>

#define T_ 128
#define R_ 9
#define D_ 512
#define B_ 16

typedef __attribute__((ext_vector_type(16))) __bf16 bf16x16;
typedef __attribute__((ext_vector_type(8)))  float  f32x8;

union FragU { bf16x16 f; uint4 q[2]; };

__device__ __forceinline__ unsigned short bf16_rne(float x) {
    unsigned u = __builtin_bit_cast(unsigned, x);
    u += 0x7FFFu + ((u >> 16) & 1u);           // round-to-nearest-even
    return (unsigned short)(u >> 16);
}
__device__ __forceinline__ float bf16_to_f32(unsigned short h) {
    unsigned u = ((unsigned)h) << 16;
    return __builtin_bit_cast(float, u);
}

// ---------------------------------------------------------------------------
// Kernel 1: f2f similarity. Per block: one 144x144 tile of S_b = Q_b * T_b^T
// (rows=(i,o), cols=(j,p)), accumulated with 3-pass bf16 hi/lo WMMA, reduced
// (max over p, mean over o) to a 16x16 block of the 128x128 similarity map.
// ---------------------------------------------------------------------------
#define KSTR 40     // ushort stride of a staged 32-wide bf16 row (padded)
#define SSTR 148    // float stride of the 144x144 score tile (padded)

__global__ __launch_bounds__(288)
void f2f_wmma_kernel(const float* __restrict__ anchors,
                     const float* __restrict__ positives,
                     const float* __restrict__ negatives,
                     const float* __restrict__ qmask,
                     const float* __restrict__ pmask,
                     const float* __restrict__ nmask,
                     float* __restrict__ simred)
{
    __shared__ union {
        struct {
            unsigned short ah[144 * KSTR];
            unsigned short al[144 * KSTR];
            unsigned short bh[144 * KSTR];
            unsigned short bl[144 * KSTR];
        } st;
        float s[144 * SSTR];
    } lds;

    const int blk  = blockIdx.x;
    const int pair = blk >> 10;          // 0: anchor x positive, 1: anchor x negative
    const int rem  = blk & 1023;
    const int b    = rem >> 6;
    const int ti   = (rem >> 3) & 7;     // 8 row tiles of 144
    const int tj   = rem & 7;            // 8 col tiles of 144

    const float* tsrc = pair ? negatives : positives;
    const float* Q  = anchors + ((size_t)b * (T_ * R_) + ti * 144) * D_;
    const float* Tt = tsrc    + ((size_t)b * (T_ * R_) + tj * 144) * D_;

    const int tid  = threadIdx.x;
    const int wave = tid >> 5;           // 0..8, one 16-row strip each
    const int lane = tid & 31;
    const int m    = lane & 15;
    const int half = lane >> 4;

    f32x8 C[9];
    const f32x8 zero = {0.f, 0.f, 0.f, 0.f, 0.f, 0.f, 0.f, 0.f};
#pragma unroll
    for (int n = 0; n < 9; ++n) C[n] = zero;

    for (int k0 = 0; k0 < D_; k0 += 32) {
        // ---- stage 144x32 f32 of Q and T, split into bf16 hi/lo planes ----
        for (int idx = tid; idx < 1152; idx += 288) {     // exactly 4 iters/thread
            const int row = idx >> 3;
            const int seg = (idx & 7) << 2;
            const float4 vq = *(const float4*)(Q  + (size_t)row * D_ + k0 + seg);
            const float4 vt = *(const float4*)(Tt + (size_t)row * D_ + k0 + seg);
            if (k0 + 32 < D_) {
                __builtin_prefetch(Q  + (size_t)row * D_ + k0 + 32 + seg, 0, 0);
                __builtin_prefetch(Tt + (size_t)row * D_ + k0 + 32 + seg, 0, 0);
            }
            const float qv[4] = {vq.x, vq.y, vq.z, vq.w};
            const float tv[4] = {vt.x, vt.y, vt.z, vt.w};
#pragma unroll
            for (int e = 0; e < 4; ++e) {
                unsigned short h = bf16_rne(qv[e]);
                lds.st.ah[row * KSTR + seg + e] = h;
                lds.st.al[row * KSTR + seg + e] = bf16_rne(qv[e] - bf16_to_f32(h));
                unsigned short g = bf16_rne(tv[e]);
                lds.st.bh[row * KSTR + seg + e] = g;
                lds.st.bl[row * KSTR + seg + e] = bf16_rne(tv[e] - bf16_to_f32(g));
            }
        }
        __syncthreads();

        // ---- A fragments (16x32) for this wave's row strip ----
        FragU aH, aL;
        {
            const unsigned short* ar = &lds.st.ah[(wave * 16 + m) * KSTR + half * 16];
            aH.q[0] = *(const uint4*)(ar);
            aH.q[1] = *(const uint4*)(ar + 8);
            const unsigned short* al_ = &lds.st.al[(wave * 16 + m) * KSTR + half * 16];
            aL.q[0] = *(const uint4*)(al_);
            aL.q[1] = *(const uint4*)(al_ + 8);
        }
#pragma unroll
        for (int n = 0; n < 9; ++n) {
            FragU bH, bL;
            const unsigned short* br  = &lds.st.bh[(n * 16 + m) * KSTR + half * 16];
            bH.q[0] = *(const uint4*)(br);
            bH.q[1] = *(const uint4*)(br + 8);
            const unsigned short* bl_ = &lds.st.bl[(n * 16 + m) * KSTR + half * 16];
            bL.q[0] = *(const uint4*)(bl_);
            bL.q[1] = *(const uint4*)(bl_ + 8);
            // q*t ~= qh*th + qh*tl + ql*th  (drops ~2^-16 ql*tl term)
            C[n] = __builtin_amdgcn_wmma_f32_16x16x32_bf16(false, aH.f, false, bH.f,
                                                           (short)0, C[n], false, false);
            C[n] = __builtin_amdgcn_wmma_f32_16x16x32_bf16(false, aH.f, false, bL.f,
                                                           (short)0, C[n], false, false);
            C[n] = __builtin_amdgcn_wmma_f32_16x16x32_bf16(false, aL.f, false, bH.f,
                                                           (short)0, C[n], false, false);
        }
        __syncthreads();
    }

    // ---- dump 144x144 scores, then reduce to 16x16 (max over p, mean over o) ----
#pragma unroll
    for (int n = 0; n < 9; ++n) {
#pragma unroll
        for (int v = 0; v < 8; ++v) {
            const int rl = wave * 16 + half * 8 + v;   // C/D layout: M = v + 8*half
            lds.s[rl * SSTR + n * 16 + m] = C[n][v];
        }
    }
    __syncthreads();

    if (tid < 256) {
        const int iL = tid >> 4, jL = tid & 15;
        float acc = 0.f;
#pragma unroll
        for (int o = 0; o < 9; ++o) {
            const float* row = &lds.s[(iL * 9 + o) * SSTR + jL * 9];
            float best = row[0];
#pragma unroll
            for (int p = 1; p < 9; ++p) best = fmaxf(best, row[p]);
            acc += best;
        }
        float val = acc * (1.0f / 9.0f);
        const int gi = ti * 16 + iL, gj = tj * 16 + jL;
        const float* tm = pair ? nmask : pmask;
        if (qmask[b * T_ + gi] * tm[b * T_ + gj] <= 0.f) val = 0.f;
        simred[(((size_t)pair * B_ + b) * T_ + gi) * T_ + gj] = val;
    }
}

// ---------------------------------------------------------------------------
// Kernel 2: conv1(1->32, 3x3, pad1) + relu + maxpool2 : (32,1,128,128)->(32,32,64,64)
// ---------------------------------------------------------------------------
__global__ __launch_bounds__(256)
void conv1_pool_kernel(const float* __restrict__ simred,
                       const float* __restrict__ w, const float* __restrict__ bias,
                       float* __restrict__ out)
{
    const int idx = blockIdx.x * blockDim.x + threadIdx.x;
    if (idx >= 32 * 32 * 64 * 64) return;
    const int x = idx & 63, y = (idx >> 6) & 63, c = (idx >> 12) & 31, b = idx >> 17;
    const float* in = simred + (size_t)b * (128 * 128);
    float wv[9];
#pragma unroll
    for (int k = 0; k < 9; ++k) wv[k] = w[c * 9 + k];
    const float bv = bias[c];
    float best = 0.f;
#pragma unroll
    for (int dy = 0; dy < 2; ++dy)
#pragma unroll
    for (int dx = 0; dx < 2; ++dx) {
        const int cy = 2 * y + dy, cx = 2 * x + dx;
        float acc = bv;
#pragma unroll
        for (int ky = 0; ky < 3; ++ky) {
            const int sy = cy + ky - 1;
            if ((unsigned)sy >= 128u) continue;
#pragma unroll
            for (int kx = 0; kx < 3; ++kx) {
                const int sx = cx + kx - 1;
                if ((unsigned)sx >= 128u) continue;
                acc = fmaf(wv[ky * 3 + kx], in[sy * 128 + sx], acc);
            }
        }
        best = fmaxf(best, fmaxf(acc, 0.f));
    }
    out[idx] = best;
}

// ---------------------------------------------------------------------------
// Kernel 3: conv2(32->64, 3x3, pad1) + relu + maxpool2 : (32,32,64,64)->(32,64,32,32)
// ---------------------------------------------------------------------------
__global__ __launch_bounds__(256)
void conv2_pool_kernel(const float* __restrict__ x1p, const float* __restrict__ w,
                       const float* __restrict__ bias, float* __restrict__ out)
{
    const int idx = blockIdx.x * blockDim.x + threadIdx.x;
    if (idx >= 32 * 64 * 32 * 32) return;
    const int x = idx & 31, y = (idx >> 5) & 31, c = (idx >> 10) & 63, b = idx >> 16;
    const float* wc = w + (size_t)c * 32 * 9;
    const float bv = bias[c];
    float best = 0.f;
#pragma unroll
    for (int dy = 0; dy < 2; ++dy)
#pragma unroll
    for (int dx = 0; dx < 2; ++dx) {
        const int cy = 2 * y + dy, cx = 2 * x + dx;
        float acc = bv;
        for (int ic = 0; ic < 32; ++ic) {
            const float* in = x1p + (size_t)(b * 32 + ic) * 64 * 64;
            const float* wk = wc + ic * 9;
#pragma unroll
            for (int ky = 0; ky < 3; ++ky) {
                const int sy = cy + ky - 1;
                if ((unsigned)sy >= 64u) continue;
#pragma unroll
                for (int kx = 0; kx < 3; ++kx) {
                    const int sx = cx + kx - 1;
                    if ((unsigned)sx >= 64u) continue;
                    acc = fmaf(wk[ky * 3 + kx], in[sy * 64 + sx], acc);
                }
            }
        }
        best = fmaxf(best, fmaxf(acc, 0.f));
    }
    out[idx] = best;
}

// ---------------------------------------------------------------------------
// Kernel 4: conv3(64->128, 3x3, pad1) + relu fused with 1x1 fconv(128->1).
// One block per output pixel, one thread per conv3 channel; deterministic
// LDS reduction (no float atomics -> bit-identical replays).
// ---------------------------------------------------------------------------
__global__ __launch_bounds__(128)
void conv3_fconv_kernel(const float* __restrict__ x2p, const float* __restrict__ w3,
                        const float* __restrict__ b3, const float* __restrict__ fw,
                        const float* __restrict__ fb, float* __restrict__ x4)
{
    const int pix = blockIdx.x;                 // (b*32 + y)*32 + x
    const int x = pix & 31, y = (pix >> 5) & 31, b = pix >> 10;
    const int oc = threadIdx.x;                 // 0..127
    float acc = b3[oc];
    const float* wb = w3 + (size_t)oc * 64 * 9;
    for (int ic = 0; ic < 64; ++ic) {
        const float* in = x2p + (size_t)(b * 64 + ic) * 32 * 32;
        const float* wk = wb + ic * 9;
#pragma unroll
        for (int ky = 0; ky < 3; ++ky) {
            const int sy = y + ky - 1;
            if ((unsigned)sy >= 32u) continue;
#pragma unroll
            for (int kx = 0; kx < 3; ++kx) {
                const int sx = x + kx - 1;
                if ((unsigned)sx >= 32u) continue;
                acc = fmaf(wk[ky * 3 + kx], in[sy * 32 + sx], acc);
            }
        }
    }
    __shared__ float red[128];
    red[oc] = fmaxf(acc, 0.f) * fw[oc];
    __syncthreads();
    for (int s = 64; s > 0; s >>= 1) {
        if (oc < s) red[oc] += red[oc + s];
        __syncthreads();
    }
    if (oc == 0) x4[pix] = red[0] + fb[0];
}

// ---------------------------------------------------------------------------
// Kernel 5: loss + clip + v2v similarity -> d_out[0..15]=pos, [16..31]=neg, [32]=loss
// ---------------------------------------------------------------------------
__global__ __launch_bounds__(256)
void finalize_kernel(const float* __restrict__ x4, const float* __restrict__ qmask,
                     const float* __restrict__ pmask, const float* __restrict__ nmask,
                     float* __restrict__ outp)
{
    const int tid = threadIdx.x;
    __shared__ float red[256];
    __shared__ float mi[1024];
    __shared__ float mm[1024];

    // hinge loss over all head outputs (pre-clip)
    float part = 0.f;
    for (int idx = tid; idx < 32 * 32 * 32; idx += 256) {
        const float s = x4[idx];
        part += fmaxf(-1.f - s, 0.f) + fmaxf(s - 1.f, 0.f);
    }
    red[tid] = part;
    __syncthreads();
    for (int s = 128; s > 0; s >>= 1) {
        if (tid < s) red[tid] += red[tid + s];
        __syncthreads();
    }
    const float loss = red[0];

    // per (pairing-batch, row): masked row-max of clipped sim + mask row-max
    for (int idx = tid; idx < 1024; idx += 256) {
        const int bp = idx >> 5, i = idx & 31;
        const int bb = bp & 15;
        const float* tmask = (bp < 16) ? pmask : nmask;
        const float* s4 = x4 + bp * 1024 + i * 32;
        float qv[4];
#pragma unroll
        for (int d = 0; d < 4; ++d) qv[d] = qmask[bb * T_ + 4 * i + d];
        float mmax = -INFINITY, rmax = -INFINITY;
        for (int J = 0; J < 32; ++J) {
            float mp = -INFINITY;  // 4x4 maxpooled mask (qm_i * tm_j outer product)
#pragma unroll
            for (int dj = 0; dj < 4; ++dj) {
                const float tv = tmask[bb * T_ + 4 * J + dj];
#pragma unroll
                for (int di = 0; di < 4; ++di) mp = fmaxf(mp, qv[di] * tv);
            }
            mmax = fmaxf(mmax, mp);
            if (mp > 0.f) {
                float s = s4[J];
                s = fminf(fmaxf(s, -1.f), 1.f);
                rmax = fmaxf(rmax, s);
            }
        }
        mi[idx] = (mmax > 0.f) ? rmax : 0.f;
        mm[idx] = mmax;
    }
    __syncthreads();

    if (tid < 32) {
        float sn = 0.f, sd = 0.f;
        for (int i = 0; i < 32; ++i) { sn += mi[tid * 32 + i]; sd += mm[tid * 32 + i]; }
        outp[tid] = sn / sd;   // [0..15] pos_pair, [16..31] neg_pair
    }
    if (tid == 0) outp[32] = loss;
}

// ---------------------------------------------------------------------------
extern "C" void kernel_launch(void* const* d_in, const int* in_sizes, int n_in,
                              void* d_out, int out_size, void* d_ws, size_t ws_size,
                              hipStream_t stream)
{
    (void)in_sizes; (void)n_in; (void)out_size; (void)ws_size;
    const float* anchors   = (const float*)d_in[0];
    const float* positives = (const float*)d_in[1];
    const float* negatives = (const float*)d_in[2];
    const float* qmask     = (const float*)d_in[3];
    const float* pmask     = (const float*)d_in[4];
    const float* nmask     = (const float*)d_in[5];
    const float* c1w = (const float*)d_in[6];
    const float* c1b = (const float*)d_in[7];
    const float* c2w = (const float*)d_in[8];
    const float* c2b = (const float*)d_in[9];
    const float* c3w = (const float*)d_in[10];
    const float* c3b = (const float*)d_in[11];
    const float* fw  = (const float*)d_in[12];
    const float* fb  = (const float*)d_in[13];

    char* ws = (char*)d_ws;
    float* simred = (float*)(ws);                                        //  2 MB: (32,128,128)
    float* x1p    = (float*)(ws + (2u << 20));                           // 16 MB: (32,32,64,64)
    float* x2p    = (float*)(ws + (2u << 20) + (16u << 20));             //  8 MB: (32,64,32,32)
    float* x4     = (float*)(ws + (2u << 20) + (16u << 20) + (8u << 20));// 128KB: (32,32,32)

    f2f_wmma_kernel<<<2048, 288, 0, stream>>>(anchors, positives, negatives,
                                              qmask, pmask, nmask, simred);
    conv1_pool_kernel<<<(32 * 32 * 64 * 64) / 256, 256, 0, stream>>>(simred, c1w, c1b, x1p);
    conv2_pool_kernel<<<(32 * 64 * 32 * 32) / 256, 256, 0, stream>>>(x1p, c2w, c2b, x2p);
    conv3_fconv_kernel<<<32 * 32 * 32, 128, 0, stream>>>(x2p, c3w, c3b, fw, fb, x4);
    finalize_kernel<<<1, 256, 0, stream>>>(x4, qmask, pmask, nmask, (float*)d_out);
}
